// GNN_85950885527609
// MI455X (gfx1250) — matrix operand order repaired and checked by reading, not compiled, not run on hardware
//
#include <hip/hip_runtime.h>

typedef __attribute__((ext_vector_type(16))) _Float16 v16h;
typedef __attribute__((ext_vector_type(8)))  _Float16 v8h;
typedef __attribute__((ext_vector_type(4)))  _Float16 v4h;
typedef __attribute__((ext_vector_type(8)))  float    v8f;

static constexpr int IN_F = 128;
static constexpr int HID  = 256;
static constexpr int NGR  = 64;
static constexpr int OUTF = 10;

// ---------------- utility kernels ----------------

__global__ void k_zero_f32(float* __restrict__ p, long long n) {
  long long i = (long long)blockIdx.x * blockDim.x + threadIdx.x;
  long long stride = (long long)gridDim.x * blockDim.x;
  for (; i < n; i += stride) p[i] = 0.0f;
}

__global__ void k_cvt_f16(const float* __restrict__ in, _Float16* __restrict__ out, long long n) {
  long long i = (long long)blockIdx.x * blockDim.x + threadIdx.x;
  if (i < n) out[i] = (_Float16)in[i];
}

// degree of dst (self-loop added later as +1)
__global__ void k_deg(const long long* __restrict__ dst, float* __restrict__ deg, long long E) {
  long long e = (long long)blockIdx.x * blockDim.x + threadIdx.x;
  if (e < E) atomicAdd(&deg[(int)dst[e]], 1.0f);
}

__global__ void k_dinv(const float* __restrict__ deg, float* __restrict__ dinv, long long N) {
  long long i = (long long)blockIdx.x * blockDim.x + threadIdx.x;
  if (i < N) {
    float d = deg[i] + 1.0f;  // + self loop
    dinv[i] = (d > 0.0f) ? rsqrtf(d) : 0.0f;
  }
}

// ---------------- WMMA GEMM:  C[M,HID](f16) = A[M,K](f16) @ W[HID,K]^T ----------------
// block = 256 threads = 8 waves, one 16-row M-tile per block, each wave does 2 N-tiles.
// A tile (16 x K halves) staged in LDS, shared by all 8 waves.

template <int K>
__global__ __launch_bounds__(256) void k_gemm_f16(const _Float16* __restrict__ A,
                                                  const _Float16* __restrict__ W,
                                                  _Float16* __restrict__ C, int M) {
  __shared__ _Float16 As[16 * K];

  const int tid  = threadIdx.x;
  const int m0   = blockIdx.x * 16;

  // cooperative A-tile load (v8h = 16B chunks)
  const int chunks = 16 * K / 8;
  for (int c = tid; c < chunks; c += 256) {
    int row = c / (K / 8);
    int kk  = (c % (K / 8)) * 8;
    int gr  = m0 + row; if (gr >= M) gr = M - 1;
    *(v8h*)&As[row * K + kk] = *(const v8h*)(A + (size_t)gr * K + kk);
  }
  __syncthreads();

  const int lane = tid & 31;
  const int wave = tid >> 5;
  const int hi   = lane >> 4;   // 0: K low half, 1: K high half (contiguous k order, same in A and B)
  const int lr   = lane & 15;
  const int n0   = wave * 32;   // 8 waves x 32 cols = 256 = HID

  v8f acc0 = {};
  v8f acc1 = {};

#pragma unroll
  for (int k0 = 0; k0 < K; k0 += 32) {
    // A fragment: lane lr(+hi) holds row m0+lr, K = {k0+hi*16..+15}
    union { v16h v; v8h h[2]; } a;
    a.h[0] = *(const v8h*)&As[lr * K + k0 + hi * 16];
    a.h[1] = *(const v8h*)&As[lr * K + k0 + hi * 16 + 8];

    // B fragments: lane holds 16 contiguous K-halves of W row (n0+lr), half picked by hi
    union { v16h v; v8h h[2]; } b0, b1;
    const _Float16* wp0 = W + (size_t)(n0 + lr) * K + k0 + hi * 16;
    b0.h[0] = *(const v8h*)wp0;
    b0.h[1] = *(const v8h*)(wp0 + 8);
    const _Float16* wp1 = W + (size_t)(n0 + 16 + lr) * K + k0 + hi * 16;
    b1.h[0] = *(const v8h*)wp1;
    b1.h[1] = *(const v8h*)(wp1 + 8);

    acc0 = __builtin_amdgcn_wmma_f32_16x16x32_f16(false, a.v, false, b0.v, (short)0, acc0, false, false);
    acc1 = __builtin_amdgcn_wmma_f32_16x16x32_f16(false, a.v, false, b1.v, (short)0, acc1, false, false);
  }

  // D layout: VGPR r -> row m0 + hi*8 + r, col n0 + lr
#pragma unroll
  for (int r = 0; r < 8; ++r) {
    int row = m0 + hi * 8 + r;
    if (row < M) {
      C[(size_t)row * HID + n0 + lr]      = (_Float16)acc0[r];
      C[(size_t)row * HID + n0 + 16 + lr] = (_Float16)acc1[r];
    }
  }
}

// ---------------- edge scatter: agg[dst] += dinv[s]*dinv[d] * h[src] ----------------
// 64 threads per edge, 4 features per thread (v4h load, 4 f32 atomics; L2-resident).

__global__ void k_scatter(const long long* __restrict__ src, const long long* __restrict__ dst,
                          const float* __restrict__ dinv, const _Float16* __restrict__ h,
                          float* __restrict__ agg, long long E) {
  long long t = (long long)blockIdx.x * blockDim.x + threadIdx.x;
  if (t >= E * 64) return;
  int e = (int)(t >> 6);
  int f = (int)(t & 63) * 4;
  int s = (int)src[e];
  int d = (int)dst[e];
  float nm = dinv[s] * dinv[d];
  v4h hv = *(const v4h*)(h + (size_t)s * HID + f);
  float* ap = agg + (size_t)d * HID + f;
  atomicAdd(ap + 0, nm * (float)hv[0]);
  atomicAdd(ap + 1, nm * (float)hv[1]);
  atomicAdd(ap + 2, nm * (float)hv[2]);
  atomicAdd(ap + 3, nm * (float)hv[3]);
}

// ---------------- fused self-loop + bias + relu (layer 1 -> f16 features) ----------------

__global__ void k_bias_relu(const float* __restrict__ agg, const _Float16* __restrict__ h,
                            const float* __restrict__ dinv, const float* __restrict__ b,
                            _Float16* __restrict__ out, long long total) {
  long long t = (long long)blockIdx.x * blockDim.x + threadIdx.x;
  if (t >= total) return;
  int i = (int)(t >> 8);
  int f = (int)(t & 255);
  float di = dinv[i];
  float v = agg[t] + di * di * (float)h[t] + b[f];
  out[t] = (_Float16)fmaxf(v, 0.0f);
}

// ---------------- fused self-loop + bias + relu + mean-pool accumulation (layer 2) ----------------

__global__ void k_bias_relu_pool(const float* __restrict__ agg, const _Float16* __restrict__ h,
                                 const float* __restrict__ dinv, const float* __restrict__ b,
                                 const long long* __restrict__ batch, float* __restrict__ pooled,
                                 long long total) {
  long long t = (long long)blockIdx.x * blockDim.x + threadIdx.x;
  if (t >= total) return;
  int i = (int)(t >> 8);
  int f = (int)(t & 255);
  float di = dinv[i];
  float v = fmaxf(agg[t] + di * di * (float)h[t] + b[f], 0.0f);
  atomicAdd(&pooled[(int)batch[i] * HID + f], v);
}

__global__ void k_count(const long long* __restrict__ batch, float* __restrict__ cnt, long long N) {
  long long i = (long long)blockIdx.x * blockDim.x + threadIdx.x;
  if (i < N) atomicAdd(&cnt[(int)batch[i]], 1.0f);
}

// ---------------- final FC: out[g,o] = (pooled[g]/cnt[g]) . Wfc[o] + bfc[o] ----------------

__global__ void k_fc(const float* __restrict__ pooled, const float* __restrict__ cnt,
                     const float* __restrict__ Wfc, const float* __restrict__ bfc,
                     float* __restrict__ out) {
  int t = blockIdx.x * blockDim.x + threadIdx.x;
  if (t >= NGR * OUTF) return;
  int g = t / OUTF;
  int o = t % OUTF;
  float inv = 1.0f / fmaxf(cnt[g], 1.0f);
  float acc = 0.0f;
  for (int k = 0; k < HID; ++k) acc += pooled[g * HID + k] * Wfc[o * HID + k];
  out[t] = acc * inv + bfc[o];
}

// ---------------- launch ----------------

static inline unsigned cdiv_ll(long long a, int b) { return (unsigned)((a + b - 1) / b); }

extern "C" void kernel_launch(void* const* d_in, const int* in_sizes, int n_in,
                              void* d_out, int out_size, void* d_ws, size_t ws_size,
                              hipStream_t stream) {
  const float*     x     = (const float*)d_in[0];
  const long long* ei    = (const long long*)d_in[1];
  const long long* batch = (const long long*)d_in[2];
  const float*     W1    = (const float*)d_in[3];
  const float*     b1    = (const float*)d_in[4];
  const float*     W2    = (const float*)d_in[5];
  const float*     b2    = (const float*)d_in[6];
  const float*     Wfc   = (const float*)d_in[7];
  const float*     bfc   = (const float*)d_in[8];
  float*           out   = (float*)d_out;

  const long long N = in_sizes[0] / IN_F;
  const long long E = in_sizes[1] / 2;
  const long long* src = ei;
  const long long* dst = ei + E;

  // carve workspace
  char* ws = (char*)d_ws;
  size_t off = 0;
  auto carve = [&](size_t bytes) -> void* {
    off = (off + 255) & ~(size_t)255;
    void* p = ws + off;
    off += bytes;
    return p;
  };
  float*     deg    = (float*)carve(N * 4);
  float*     dinv   = (float*)carve(N * 4);
  _Float16*  xh     = (_Float16*)carve((size_t)N * IN_F * 2);
  _Float16*  W1h    = (_Float16*)carve((size_t)HID * IN_F * 2);
  _Float16*  W2h    = (_Float16*)carve((size_t)HID * HID * 2);
  _Float16*  h      = (_Float16*)carve((size_t)N * HID * 2);   // GEMM outputs (reused layer1/2)
  _Float16*  ah     = (_Float16*)carve((size_t)N * HID * 2);   // layer-1 activations
  float*     agg    = (float*)carve((size_t)N * HID * 4);      // scatter accumulator (reused)
  float*     pooled = (float*)carve((size_t)NGR * HID * 4);
  float*     cnt    = (float*)carve((size_t)NGR * 4);
  (void)ws_size; (void)n_in; (void)out_size;

  const int B = 256;
  const long long NH = N * HID;

  // normalization
  k_zero_f32<<<4096, B, 0, stream>>>(deg, N);
  k_zero_f32<<<256, B, 0, stream>>>(pooled, (long long)NGR * HID + NGR); // pooled+cnt contiguous
  k_deg<<<cdiv_ll(E, B), B, 0, stream>>>(dst, deg, E);
  k_dinv<<<cdiv_ll(N, B), B, 0, stream>>>(deg, dinv, N);

  // f32 -> f16 operands
  k_cvt_f16<<<cdiv_ll(N * IN_F, B), B, 0, stream>>>(x, xh, N * IN_F);
  k_cvt_f16<<<cdiv_ll((long long)HID * IN_F, B), B, 0, stream>>>(W1, W1h, (long long)HID * IN_F);
  k_cvt_f16<<<cdiv_ll((long long)HID * HID, B), B, 0, stream>>>(W2, W2h, (long long)HID * HID);

  // layer 1: GEMM -> scatter -> bias/relu
  k_gemm_f16<IN_F><<<cdiv_ll(N, 16), B, 0, stream>>>(xh, W1h, h, (int)N);
  k_zero_f32<<<8192, B, 0, stream>>>(agg, NH);
  k_scatter<<<cdiv_ll(E * 64, B), B, 0, stream>>>(src, dst, dinv, h, agg, E);
  k_bias_relu<<<cdiv_ll(NH, B), B, 0, stream>>>(agg, h, dinv, b1, ah, NH);

  // layer 2: GEMM -> scatter -> bias/relu fused with mean-pool accumulation
  k_gemm_f16<HID><<<cdiv_ll(N, 16), B, 0, stream>>>(ah, W2h, h, (int)N);
  k_zero_f32<<<8192, B, 0, stream>>>(agg, NH);
  k_scatter<<<cdiv_ll(E * 64, B), B, 0, stream>>>(src, dst, dinv, h, agg, E);
  k_count<<<cdiv_ll(N, B), B, 0, stream>>>(batch, cnt, N);
  k_bias_relu_pool<<<cdiv_ll(NH, B), B, 0, stream>>>(agg, h, dinv, b2, batch, pooled, NH);

  // final FC
  k_fc<<<cdiv_ll((long long)NGR * OUTF, B), B, 0, stream>>>(pooled, cnt, Wfc, bfc, out);
}
